// DenseSynthesizerAttention_20100446945666
// MI455X (gfx1250) — matrix operand order repaired and checked by reading, not compiled, not run on hardware
//
#include <hip/hip_runtime.h>
#include <hip/hip_bf16.h>
#include <cstdint>

typedef __attribute__((ext_vector_type(16))) __bf16 bf16x16;
typedef __attribute__((ext_vector_type(8)))  __bf16 bf16x8;
typedef __attribute__((ext_vector_type(4)))  __bf16 bf16x4;
typedef __attribute__((ext_vector_type(8)))  float  f32x8;
typedef __attribute__((ext_vector_type(4)))  int    i32x4;

union Frag16 { bf16x16 v; bf16x8 h[2]; };

__device__ __forceinline__ f32x8 wmma_bf16(const Frag16& a, const Frag16& b, f32x8 c) {
  return __builtin_amdgcn_wmma_f32_16x16x32_bf16(false, a.v, false, b.v, (short)0, c, false, false);
}

__device__ __forceinline__ f32x8 f32x8_zero() {
  f32x8 z;
#pragma unroll
  for (int e = 0; e < 8; ++e) z[e] = 0.f;
  return z;
}

// ---- gfx1250 async global->LDS copy (ASYNCcnt path), with safe fallback ----
#if defined(__gfx1250__) && __has_builtin(__builtin_amdgcn_global_load_async_to_lds_b128)
#define HAS_ASYNC_LDS 1
#else
#define HAS_ASYNC_LDS 0
#endif

typedef __attribute__((address_space(1))) i32x4* g_v4i_ptr;
typedef __attribute__((address_space(3))) i32x4* l_v4i_ptr;

__device__ __forceinline__ void async_cp16(const void* g, void* l) {
#if HAS_ASYNC_LDS
  __builtin_amdgcn_global_load_async_to_lds_b128(
      (g_v4i_ptr)(uintptr_t)g,
      (l_v4i_ptr)(uint32_t)(uintptr_t)l,
      0, 0);
#else
  *reinterpret_cast<bf16x8*>(l) = *reinterpret_cast<const bf16x8*>(g);
#endif
}

__device__ __forceinline__ void async_wait() {
#if HAS_ASYNC_LDS
#if __has_builtin(__builtin_amdgcn_s_wait_asynccnt)
  __builtin_amdgcn_s_wait_asynccnt(0);
#else
  asm volatile("s_wait_asynccnt 0x0" ::: "memory");
#endif
#endif
}

// ---------------------------------------------------------------------------
// C = act(A[M,K] @ W[K,N] + bias[N]); A is f32 or bf16, C is f32 or bf16.
// bf16 WMMA, f32 accumulate. BM=128, BN=128, BK=32, double-buffered LDS.
// 256 threads = 8 waves (4 in M x 2 in N); each wave: 32x64 = 2x4 wmma tiles.
// M,N,K multiples of 128/128/32 (true for all four GEMMs here).
// ---------------------------------------------------------------------------
template <typename TA, typename TC>
__global__ __launch_bounds__(256) void gemm_kernel(
    const TA* __restrict__ A, const float* __restrict__ W,
    const float* __restrict__ bias, TC* __restrict__ C,
    int M, int N, int K, int relu)
{
  __shared__ __bf16 As[2][128][40];   // row stride 80B (multiple of 16B)
  __shared__ __bf16 Ws[2][128][40];   // Ws[n][k] (W transposed tile)

  const int tid  = threadIdx.x;
  const int lane = tid & 31;
  const int warp = tid >> 5;
  const int l16  = lane & 15;
  const int wm   = warp >> 1;            // 0..3
  const int wn   = warp & 1;             // 0..1
  const int rowBase = blockIdx.y * 128;
  const int colBase = blockIdx.x * 128;

  auto stage = [&](int buf, int k0) {
    if constexpr (sizeof(TA) == 4) {
      // fp32 A: vector load + convert
#pragma unroll
      for (int it = 0; it < 4; ++it) {
        int idx = tid + it * 256;          // 0..1023 float4
        int r   = idx >> 3;                // 0..127
        int c4  = (idx & 7) << 2;          // 0..28
        const float4 f = *reinterpret_cast<const float4*>(
            (const float*)A + (size_t)(rowBase + r) * K + k0 + c4);
        bf16x4 t; t[0] = (__bf16)f.x; t[1] = (__bf16)f.y;
                  t[2] = (__bf16)f.z; t[3] = (__bf16)f.w;
        *reinterpret_cast<bf16x4*>(&As[buf][r][c4]) = t;
      }
    } else {
      // bf16 A: direct 16B async copies to LDS
#pragma unroll
      for (int it = 0; it < 2; ++it) {
        int idx = tid + it * 256;          // 0..511 chunks of 8 bf16
        int r   = idx >> 2;                // 0..127
        int c8  = (idx & 3) << 3;          // 0,8,16,24
        async_cp16((const __bf16*)A + (size_t)(rowBase + r) * K + k0 + c8,
                   &As[buf][r][c8]);
      }
    }
    // W tile (fp32) 32x128 transposed into Ws[n][k]
#pragma unroll
    for (int it = 0; it < 4; ++it) {
      int idx = tid + it * 256;            // 0..1023 float4
      int kr  = idx >> 5;                  // 0..31
      int n4  = (idx & 31) << 2;           // 0..124
      const float4 f = *reinterpret_cast<const float4*>(
          W + (size_t)(k0 + kr) * N + colBase + n4);
      Ws[buf][n4 + 0][kr] = (__bf16)f.x;
      Ws[buf][n4 + 1][kr] = (__bf16)f.y;
      Ws[buf][n4 + 2][kr] = (__bf16)f.z;
      Ws[buf][n4 + 3][kr] = (__bf16)f.w;
    }
  };

  f32x8 acc[2][4];
#pragma unroll
  for (int i = 0; i < 2; ++i)
#pragma unroll
    for (int j = 0; j < 4; ++j) acc[i][j] = f32x8_zero();

  const int ka = (lane < 16) ? 0 : 8;    // A-frag K base per ISA layout
  const int kb = (lane < 16) ? 0 : 8;    // B-frag K base per ISA layout
  const int nk = K >> 5;

  stage(0, 0);
  for (int ki = 0; ki < nk; ++ki) {
    const int cur = ki & 1;
    if constexpr (sizeof(TA) == 2) async_wait();
    __syncthreads();                      // staged tile 'cur' now visible
    if (ki + 1 < nk) stage(cur ^ 1, (ki + 1) << 5);  // prefetch next tile

    Frag16 a[2];
#pragma unroll
    for (int tm = 0; tm < 2; ++tm) {
      int r = wm * 32 + tm * 16 + l16;
      a[tm].h[0] = *reinterpret_cast<const bf16x8*>(&As[cur][r][ka]);
      a[tm].h[1] = *reinterpret_cast<const bf16x8*>(&As[cur][r][ka + 16]);
    }
    Frag16 b[4];
#pragma unroll
    for (int tn = 0; tn < 4; ++tn) {
      int n = wn * 64 + tn * 16 + l16;
      b[tn].h[0] = *reinterpret_cast<const bf16x8*>(&Ws[cur][n][kb]);
      b[tn].h[1] = *reinterpret_cast<const bf16x8*>(&Ws[cur][n][kb + 16]);
    }
#pragma unroll
    for (int tm = 0; tm < 2; ++tm)
#pragma unroll
      for (int tn = 0; tn < 4; ++tn)
        acc[tm][tn] = wmma_bf16(a[tm], b[tn], acc[tm][tn]);
  }

  // epilogue: bias (+relu), store as TC
#pragma unroll
  for (int tm = 0; tm < 2; ++tm)
#pragma unroll
    for (int tn = 0; tn < 4; ++tn) {
      int col = colBase + wn * 64 + tn * 16 + l16;
      float bb = bias[col];
#pragma unroll
      for (int e = 0; e < 8; ++e) {
        int row = rowBase + wm * 32 + tm * 16 + ((lane < 16) ? e : e + 8);
        float v = acc[tm][tn][e] + bb;
        if (relu) v = fmaxf(v, 0.f);
        C[(size_t)row * N + col] = (TC)v;
      }
    }
}

// ---------------------------------------------------------------------------
// Fused flash-style synthesizer attention per (b,h):
//   scores = (aw @ aw^T) / 8 ; attn = softmax(scores) ; x_h = attn @ v_h
// aw, vp are [8192,512] bf16; head slice = cols [64h,64h+64). x out is bf16.
// Block = (q-tile of 128) x (b,h). 256 threads = 8 waves; wave w owns query
// rows [16w,16w+16). Loops over 16 key tiles of 64 with online softmax.
// ---------------------------------------------------------------------------
__global__ __launch_bounds__(256) void synth_attn_kernel(
    const __bf16* __restrict__ aw, const __bf16* __restrict__ vp,
    __bf16* __restrict__ x)
{
  __shared__ __bf16 Qs[128][72];       // row stride 144B
  __shared__ __bf16 KP[128 * 72];      // K tile [64][72] overlaid by P [128][72]
  __shared__ __bf16 Vt[64][72];        // V transposed: Vt[dim][key]

  const int tid  = threadIdx.x;
  const int lane = tid & 31;
  const int warp = tid >> 5;           // 0..7
  const int l16  = lane & 15;
  const int bh   = blockIdx.y;         // 0..63
  const int b    = bh >> 3;
  const int h    = bh & 7;
  const int qBase = blockIdx.x * 128;

  const size_t base = ((size_t)b * 1024) * 512 + (size_t)h * 64;

  // ---- stage Q tile 128x64 bf16 via async 16B copies (resident) ----
#pragma unroll
  for (int it = 0; it < 4; ++it) {
    int idx = tid + it * 256;          // 0..1023 chunks of 8 bf16
    int r   = idx >> 3;                // 0..127
    int c8  = (idx & 7) << 3;          // 0..56
    async_cp16(aw + base + (size_t)(qBase + r) * 512 + c8, &Qs[r][c8]);
  }

  f32x8 mrow, lrow, o[4];
#pragma unroll
  for (int e = 0; e < 8; ++e) { mrow[e] = -3.0e38f; lrow[e] = 0.f; }
#pragma unroll
  for (int tn = 0; tn < 4; ++tn) o[tn] = f32x8_zero();

  const int ka = (lane < 16) ? 0 : 8;
  const int kb = (lane < 16) ? 0 : 8;

  for (int kt = 0; kt < 1024; kt += 64) {
    // ---- stage K tile [64 keys][64 dims] row-major into KP (async) ----
#pragma unroll
    for (int it = 0; it < 2; ++it) {
      int idx = tid + it * 256;        // 0..511 chunks
      int r   = idx >> 3;              // 0..63
      int c8  = (idx & 7) << 3;
      async_cp16(aw + base + (size_t)(kt + r) * 512 + c8, &KP[r * 72 + c8]);
    }
    // ---- stage V tile transposed: Vt[dim][key] (manual) ----
    {
      int r   = tid >> 2;              // key 0..63
      int c16 = (tid & 3) << 4;        // dim 0,16,32,48
      const __bf16* gp = vp + base + (size_t)(kt + r) * 512 + c16;
      const bf16x8 v0 = *reinterpret_cast<const bf16x8*>(gp);
      const bf16x8 v1 = *reinterpret_cast<const bf16x8*>(gp + 8);
#pragma unroll
      for (int j = 0; j < 8; ++j) {
        Vt[c16 + j][r]     = v0[j];
        Vt[c16 + 8 + j][r] = v1[j];
      }
    }
    async_wait();
    __syncthreads();

    // ---- S = Q @ K^T for this wave's 16 rows x 64 keys ----
    f32x8 s[4];
#pragma unroll
    for (int tn = 0; tn < 4; ++tn) s[tn] = f32x8_zero();
#pragma unroll
    for (int ks = 0; ks < 64; ks += 32) {
      Frag16 af;
      int r = warp * 16 + l16;
      af.h[0] = *reinterpret_cast<const bf16x8*>(&Qs[r][ks + ka]);
      af.h[1] = *reinterpret_cast<const bf16x8*>(&Qs[r][ks + ka + 16]);
#pragma unroll
      for (int tn = 0; tn < 4; ++tn) {
        Frag16 bfr;
        int n = tn * 16 + l16;
        bfr.h[0] = *reinterpret_cast<const bf16x8*>(&KP[n * 72 + ks + kb]);
        bfr.h[1] = *reinterpret_cast<const bf16x8*>(&KP[n * 72 + ks + kb + 16]);
        s[tn] = wmma_bf16(af, bfr, s[tn]);
      }
    }

    // ---- online softmax update (1/sqrt(64) = 0.125 scale) ----
    f32x8 mt;
#pragma unroll
    for (int e = 0; e < 8; ++e) {
      float v0 = s[0][e] * 0.125f, v1 = s[1][e] * 0.125f;
      float v2 = s[2][e] * 0.125f, v3 = s[3][e] * 0.125f;
      s[0][e] = v0; s[1][e] = v1; s[2][e] = v2; s[3][e] = v3;
      mt[e] = fmaxf(fmaxf(v0, v1), fmaxf(v2, v3));
    }
#pragma unroll
    for (int mask = 1; mask < 16; mask <<= 1)
#pragma unroll
      for (int e = 0; e < 8; ++e) mt[e] = fmaxf(mt[e], __shfl_xor(mt[e], mask, 32));

    f32x8 sc, ls;
#pragma unroll
    for (int e = 0; e < 8; ++e) {
      float mn = fmaxf(mrow[e], mt[e]);
      sc[e]   = __expf(mrow[e] - mn);
      mrow[e] = mn;
      float p0 = __expf(s[0][e] - mn); float p1 = __expf(s[1][e] - mn);
      float p2 = __expf(s[2][e] - mn); float p3 = __expf(s[3][e] - mn);
      s[0][e] = p0; s[1][e] = p1; s[2][e] = p2; s[3][e] = p3;
      ls[e] = p0 + p1 + p2 + p3;
    }
#pragma unroll
    for (int mask = 1; mask < 16; mask <<= 1)
#pragma unroll
      for (int e = 0; e < 8; ++e) ls[e] += __shfl_xor(ls[e], mask, 32);
#pragma unroll
    for (int e = 0; e < 8; ++e) lrow[e] = lrow[e] * sc[e] + ls[e];
#pragma unroll
    for (int tn = 0; tn < 4; ++tn)
#pragma unroll
      for (int e = 0; e < 8; ++e) o[tn][e] *= sc[e];

    __syncthreads();  // all waves done reading K tile before P overlays it

    // ---- write P (bf16) for this wave's rows; same-wave LDS RAW is in-order
#pragma unroll
    for (int tn = 0; tn < 4; ++tn)
#pragma unroll
      for (int e = 0; e < 8; ++e) {
        int r = warp * 16 + ((lane < 16) ? e : e + 8);
        int c = tn * 16 + l16;
        KP[r * 72 + c] = (__bf16)s[tn][e];
      }

    // ---- O += P @ V  (K = 64 keys, N = 64 dims) ----
#pragma unroll
    for (int ks = 0; ks < 64; ks += 32) {
      Frag16 af;
      int r = warp * 16 + l16;
      af.h[0] = *reinterpret_cast<const bf16x8*>(&KP[r * 72 + ks + ka]);
      af.h[1] = *reinterpret_cast<const bf16x8*>(&KP[r * 72 + ks + ka + 16]);
#pragma unroll
      for (int tn = 0; tn < 4; ++tn) {
        Frag16 bfr;
        int n = tn * 16 + l16;
        bfr.h[0] = *reinterpret_cast<const bf16x8*>(&Vt[n][ks + kb]);
        bfr.h[1] = *reinterpret_cast<const bf16x8*>(&Vt[n][ks + kb + 16]);
        o[tn] = wmma_bf16(af, bfr, o[tn]);
      }
    }
    __syncthreads();  // before next iteration overwrites KP / Vt
  }

  // ---- normalize and write merged [B,S,H*dk] bf16 output slice ----
#pragma unroll
  for (int tn = 0; tn < 4; ++tn) {
    int col = h * 64 + tn * 16 + l16;
#pragma unroll
    for (int e = 0; e < 8; ++e) {
      int row = qBase + warp * 16 + ((lane < 16) ? e : e + 8);
      x[((size_t)(b * 1024 + row)) * 512 + col] = (__bf16)(o[tn][e] / lrow[e]);
    }
  }
}

extern "C" void kernel_launch(void* const* d_in, const int* in_sizes, int n_in,
                              void* d_out, int out_size, void* d_ws, size_t ws_size,
                              hipStream_t stream) {
  (void)in_sizes; (void)n_in; (void)out_size; (void)ws_size;
  const float* query = (const float*)d_in[0];
  /* d_in[1] = key_t : unused by synthesizer attention */
  const float* value = (const float*)d_in[2];
  const float* w1 = (const float*)d_in[3];
  const float* b1 = (const float*)d_in[4];
  const float* w2 = (const float*)d_in[5];
  const float* b2 = (const float*)d_in[6];
  const float* wv = (const float*)d_in[7];
  const float* bvec = (const float*)d_in[8];
  const float* wo = (const float*)d_in[9];
  const float* bo = (const float*)d_in[10];
  float* out = (float*)d_out;

  // bf16 intermediates in workspace: 16 + 8 + 8 + 8 = 40 MB
  __bf16* ws   = (__bf16*)d_ws;
  __bf16* hbuf = ws;                                   // 8192*1024 bf16
  __bf16* awb  = hbuf + (size_t)8192 * 1024;           // 8192*512 bf16
  __bf16* vpb  = awb  + (size_t)8192 * 512;
  __bf16* xb   = vpb  + (size_t)8192 * 512;

  dim3 blk(256);
  // h = relu(Q @ w1 + b1)            [8192,512]x[512,1024] -> bf16
  gemm_kernel<float, __bf16><<<dim3(8, 64), blk, 0, stream>>>(query, w1, b1, hbuf, 8192, 1024, 512, 1);
  // aw = h @ w2 + b2                 [8192,1024]x[1024,512] -> bf16
  gemm_kernel<__bf16, __bf16><<<dim3(4, 64), blk, 0, stream>>>(hbuf, w2, b2, awb, 8192, 512, 1024, 0);
  // v = value @ wv + bv              [8192,512]x[512,512] -> bf16
  gemm_kernel<float, __bf16><<<dim3(4, 64), blk, 0, stream>>>(value, wv, bvec, vpb, 8192, 512, 512, 0);
  // fused softmax(aw aw^T / 8) @ v  -> x (merged head layout, bf16)
  synth_attn_kernel<<<dim3(8, 64), blk, 0, stream>>>(awb, vpb, xb);
  // out = x @ wo + bo                [8192,512]x[512,512] -> f32
  gemm_kernel<__bf16, float><<<dim3(4, 64), blk, 0, stream>>>(xb, wo, bo, out, 8192, 512, 512, 0);
}
